// TorchKNNExact_51591147159703
// MI455X (gfx1250) — compile-verified, matrix operand order checked by reading, not run on hardware
//
#include <hip/hip_runtime.h>
#include <hip/hip_bf16.h>

typedef __attribute__((ext_vector_type(16))) _Float16 v16h;
typedef __attribute__((ext_vector_type(8)))  float    v8f;

#define MM 64
#define NN 2048
#define DD 256
#define LL 64
#define JJ 384
#define NEGV -999999.0f
#define ROWS (MM*NN)          // 131072 flattened (m,n) rows
#define KBLOCKS 12            // 384 / 32
#define NTILES 4              // 64 / 16
#define GP_HALVES (NTILES*KBLOCKS*32*16)   // 24576 f16 = 49152 bytes

// ---------------- Kernel A1: lgamma table + G[j][l] = P(128+j, buckets[l]/8) ----------
__device__ float gammainc_P(float a, float x) {
    if (x <= 0.0f) return 0.0f;
    float lpre = -x + a * __logf(x) - lgammaf(a);
    if (x < a + 1.0f) {                    // series for P
        float ap = a, sum = 1.0f / a, term = sum;
        for (int k = 0; k < 500; ++k) {
            ap += 1.0f;
            term *= x / ap;
            sum += term;
            if (term < sum * 1e-8f) break;
        }
        return sum * __expf(lpre);
    } else {                               // Lentz continued fraction for Q
        float b = x + 1.0f - a, c = 1e30f, d = 1.0f / b, h = d;
        for (int k = 1; k <= 500; ++k) {
            float an = -(float)k * ((float)k - a);
            b += 2.0f;
            d = an * d + b; if (fabsf(d) < 1e-30f) d = 1e-30f;
            c = b + an / c; if (fabsf(c) < 1e-30f) c = 1e-30f;
            d = 1.0f / d;
            float del = d * c;
            h *= del;
            if (fabsf(del - 1.0f) < 1e-8f) break;
        }
        return 1.0f - __expf(lpre) * h;
    }
}

__global__ __launch_bounds__(256) void prep_kernel(const float* __restrict__ buckets,
                                                   float* __restrict__ lgtab,
                                                   float* __restrict__ G) {
    int tid = blockIdx.x * blockDim.x + threadIdx.x;
    if (tid < JJ) lgtab[tid] = lgammaf((float)(tid + 1));
    if (tid < JJ * LL) {
        int j = tid >> 6, l = tid & 63;
        G[tid] = gammainc_P(128.0f + (float)j, buckets[l] * 0.125f);
    }
}

// ---------------- Kernel A2: pack G into WMMA B-fragment layout (f16) -----------------
// B 32x16 (KxN) 16-bit layout: lanes 0-15 -> N=lane, elems K = 0..7,16..23
//                              lanes 16-31 -> N=lane-16, elems K = 8..15,24..31
__global__ __launch_bounds__(256) void pack_kernel(const float* __restrict__ G,
                                                   _Float16* __restrict__ Gpack) {
    int tid = blockIdx.x * blockDim.x + threadIdx.x;  // 4*12*32 = 1536 fragment lanes
    if (tid >= NTILES * KBLOCKS * 32) return;
    int nt   = tid / (KBLOCKS * 32);
    int kb   = (tid >> 5) % KBLOCKS;
    int lane = tid & 31;
    int col  = nt * 16 + (lane & 15);
    int kofs = (lane >= 16) ? 8 : 0;
#pragma unroll
    for (int i = 0; i < 16; ++i) {
        int k = kb * 32 + kofs + ((i & 8) << 1) + (i & 7);
        Gpack[(size_t)tid * 16 + i] = (_Float16)G[k * LL + col];
    }
}

// ---------------- Kernel B: squared distances ----------------------------------------
__global__ __launch_bounds__(256) void dists_kernel(const float* __restrict__ X,
                                                    const float* __restrict__ T,
                                                    float* __restrict__ dists) {
    int tid = blockIdx.x * blockDim.x + threadIdx.x;  // 131072
    int m = tid >> 11, n = tid & (NN - 1);
    const float4* xr = (const float4*)(X + (size_t)m * DD);
    const float4* tr = (const float4*)(T + (size_t)n * DD);
    float acc = 0.0f;
#pragma unroll 4
    for (int d = 0; d < DD / 4; ++d) {
        float4 xv = xr[d], tv = tr[d];
        float dx = xv.x - tv.x, dy = xv.y - tv.y, dz = xv.z - tv.z, dw = xv.w - tv.w;
        acc = fmaf(dx, dx, acc);
        acc = fmaf(dy, dy, acc);
        acc = fmaf(dz, dz, acc);
        acc = fmaf(dw, dw, acc);
    }
    dists[tid] = acc;   // >= 0 by construction (matches clamp)
}

// ---------------- Kernel C: WMMA einsum + p_mat / log(1-F) ----------------------------
// One wave per 16-row tile. A fragment (w Poisson weights) computed via v_exp_f32
// directly in the 16-bit A layout. The shared B operand (Gpack, 48KB) is staged into
// LDS once per block via CDNA5 async global->LDS copies (ASYNCcnt), then the K-loop
// feeds the WMMAs from ds_load_b128 instead of stalling on global loads.
__global__ __launch_bounds__(128) void knn_wmma_kernel(const float* __restrict__ dists,
                                                       const _Float16* __restrict__ Gpack,
                                                       const float* __restrict__ lgtab_g,
                                                       float* __restrict__ PM,
                                                       float* __restrict__ LP) {
    __shared__ __align__(32) _Float16 GpL[GP_HALVES];   // 49152 B
    __shared__ float lgs[JJ];
    __shared__ float Ft[4][16][65];   // per-wave F tile, padded stride

    // --- async stage Gpack -> LDS (3072 x b128 = 49152 B, 24 per thread, no divergence)
    {
        const char* gsrc = (const char*)Gpack;
        unsigned ldsbase = (unsigned)(uintptr_t)(&GpL[0]);
        for (int i = threadIdx.x; i < GP_HALVES / 8; i += 128) {
            unsigned    ldsa = ldsbase + (unsigned)i * 16u;
            const char* ga   = gsrc + (size_t)i * 16u;
            asm volatile("global_load_async_to_lds_b128 %0, %1, off"
                         :: "v"(ldsa), "v"(ga)
                         : "memory");
        }
        asm volatile("s_wait_asynccnt 0" ::: "memory");
    }
    for (int t = threadIdx.x; t < JJ; t += 128) lgs[t] = lgtab_g[t];
    __syncthreads();

    const int lane = threadIdx.x & 31;
    const int wv   = threadIdx.x >> 5;
    const int row0 = (blockIdx.x * 4 + wv) * 16;
    const int rA   = row0 + (lane & 15);

    const float hnc = dists[rA] * 0.125f;     // dists / (2*sigma2)
    const float lh  = __logf(hnc);

    v8f acc0 = {}, acc1 = {}, acc2 = {}, acc3 = {};
    const int kofs = (lane >= 16) ? 8 : 0;
    const v16h* gl = (const v16h*)GpL;

    for (int kb = 0; kb < KBLOCKS; ++kb) {
        v16h a;
#pragma unroll
        for (int i = 0; i < 16; ++i) {
            int j = kb * 32 + kofs + ((i & 8) << 1) + (i & 7);
            float w = __expf(fmaf((float)j, lh, -hnc) - lgs[j]);
            a[i] = (_Float16)w;
        }
        v16h b0 = gl[(0 * KBLOCKS + kb) * 32 + lane];
        v16h b1 = gl[(1 * KBLOCKS + kb) * 32 + lane];
        v16h b2 = gl[(2 * KBLOCKS + kb) * 32 + lane];
        v16h b3 = gl[(3 * KBLOCKS + kb) * 32 + lane];
        acc0 = __builtin_amdgcn_wmma_f32_16x16x32_f16(false, a, false, b0, (short)0, acc0, false, false);
        acc1 = __builtin_amdgcn_wmma_f32_16x16x32_f16(false, a, false, b1, (short)0, acc1, false, false);
        acc2 = __builtin_amdgcn_wmma_f32_16x16x32_f16(false, a, false, b2, (short)0, acc2, false, false);
        acc3 = __builtin_amdgcn_wmma_f32_16x16x32_f16(false, a, false, b3, (short)0, acc3, false, false);
    }

    // C layout: lane<16 -> N=lane, M=v ; lane>=16 -> N=lane-16, M=v+8
    {
        const int col   = lane & 15;
        const int rbase = (lane < 16) ? 0 : 8;
#pragma unroll
        for (int v = 0; v < 8; ++v) {
            Ft[wv][rbase + v][ 0 + col] = acc0[v];
            Ft[wv][rbase + v][16 + col] = acc1[v];
            Ft[wv][rbase + v][32 + col] = acc2[v];
            Ft[wv][rbase + v][48 + col] = acc3[v];
        }
    }

    // wave-synchronous epilogue: psum[l] telescopes to 1 - F[l]
#pragma unroll 1
    for (int r = 0; r < 16; ++r) {
        const size_t grow = (size_t)(row0 + r) * LL;
        for (int li = lane; li < LL; li += 32) {
            float Fv = Ft[wv][r][li];
            float pm = (li < LL - 1) ? (Ft[wv][r][li + 1] - Fv) : (1.0f - Fv);
            float lp = __logf(fmaxf(1.0f - Fv, 1e-30f));
            PM[grow + li] = pm;
            LP[grow + li] = lp;
        }
    }
}

// ---------------- Kernel D: sequential scan over n (in-place, clamp + NaN->NEG) -------
__global__ __launch_bounds__(256) void scan_kernel(float* __restrict__ LP) {
    int tid = blockIdx.x * blockDim.x + threadIdx.x;  // 4096 = M*L
    int m = tid >> 6, l = tid & 63;
    float acc = 0.0f;
    size_t base = ((size_t)m * NN) * LL + l;
    for (int n = 0; n < NN; ++n) {
        acc += LP[base + (size_t)n * LL];
        LP[base + (size_t)n * LL] = fmaxf(acc, NEGV);  // fmaxf(NaN, NEG) == NEG
    }
}

// ---------------- Kernel E: p[m,n] = sum_l p_mat * exp(term1 + term2) ----------------
__global__ __launch_bounds__(256) void pfinal_kernel(const float* __restrict__ PM,
                                                     const float* __restrict__ LP,
                                                     float* __restrict__ P) {
    int tid = blockIdx.x * blockDim.x + threadIdx.x;  // 131072
    int m = tid >> 11, n = tid & (NN - 1);
    size_t rb    = (size_t)tid * LL;
    size_t lastb = ((size_t)m * NN + (NN - 1)) * LL;
    size_t prevb = rb - LL;
    float s = 0.0f;
#pragma unroll 4
    for (int l = 0; l < LL; ++l) {
        float t2 = LP[lastb + l] - LP[rb + l];
        float t1 = (n == 0) ? 0.0f : ((l < LL - 1) ? LP[prevb + l + 1] : NEGV);
        s = fmaf(PM[rb + l], __expf(t1 + t2), s);
    }
    P[tid] = s;
}

// ---------------- Kernel F: class reduction + normalize (deterministic LDS tree) ------
__global__ __launch_bounds__(256) void reduce_kernel(const float* __restrict__ P,
                                                     const int* __restrict__ y,
                                                     float* __restrict__ out) {
    __shared__ float s0[256], s1[256];
    int m = blockIdx.x, t = threadIdx.x;
    float a0 = 0.0f, a1 = 0.0f;
    for (int n = t; n < NN; n += 256) {
        float p = P[(size_t)m * NN + n];
        if (y[n] == 0) a0 += p; else a1 += p;
    }
    s0[t] = a0; s1[t] = a1;
    __syncthreads();
    for (int s = 128; s > 0; s >>= 1) {
        if (t < s) { s0[t] += s0[t + s]; s1[t] += s1[t + s]; }
        __syncthreads();
    }
    if (t == 0) {
        float tot = s0[0] + s1[0];
        out[m * 2 + 0] = s0[0] / tot;
        out[m * 2 + 1] = s1[0] / tot;
    }
}

extern "C" void kernel_launch(void* const* d_in, const int* in_sizes, int n_in,
                              void* d_out, int out_size, void* d_ws, size_t ws_size,
                              hipStream_t stream) {
    const float* X       = (const float*)d_in[0];
    const float* trainX  = (const float*)d_in[1];
    const float* buckets = (const float*)d_in[2];
    const int*   trainY  = (const int*)d_in[3];
    float* out = (float*)d_out;

    char* ws = (char*)d_ws;
    float*    lgtab = (float*)(ws);                                  // 384 f32
    float*    G     = (float*)(ws + 4096);                           // 384*64 f32
    _Float16* Gpack = (_Float16*)(ws + 4096 + 98304);                // 24576 f16
    float*    dists = (float*)(ws + 4096 + 98304 + 49152);           // 131072 f32
    float*    PM    = dists + ROWS;                                  // ROWS*64 f32
    float*    LP    = PM + (size_t)ROWS * LL;                        // ROWS*64 f32
    float*    Pp    = LP + (size_t)ROWS * LL;                        // ROWS f32

    prep_kernel  <<<(JJ * LL + 255) / 256, 256, 0, stream>>>(buckets, lgtab, G);
    pack_kernel  <<<6, 256, 0, stream>>>(G, Gpack);
    dists_kernel <<<ROWS / 256, 256, 0, stream>>>(X, trainX, dists);
    knn_wmma_kernel<<<ROWS / 64, 128, 0, stream>>>(dists, Gpack, lgtab, PM, LP);
    scan_kernel  <<<(MM * LL) / 256, 256, 0, stream>>>(LP);
    pfinal_kernel<<<ROWS / 256, 256, 0, stream>>>(PM, LP, Pp);
    reduce_kernel<<<MM, 256, 0, stream>>>(Pp, trainY, out);
}
